// EncoderBlock_25958782337684
// MI455X (gfx1250) — matrix operand order repaired and checked by reading, not compiled, not run on hardware
//
#include <hip/hip_runtime.h>
#include <hip/hip_bf16.h>

// ---------------- problem constants (from reference) ----------------
#define BB   2
#define SS   2048
#define DD   1024
#define HH   16
#define DKK  64
#define DFFF 4096
#define MROWS (BB * SS)   // 4096

typedef __bf16 bf16_t;
typedef __attribute__((ext_vector_type(16))) __bf16 v16bf;
typedef __attribute__((ext_vector_type(8)))  __bf16 v8bf;
typedef __attribute__((ext_vector_type(8)))  float  v8f;
typedef __attribute__((ext_vector_type(4)))  int    v4i;
typedef __attribute__((address_space(1))) v4i v4i_g;   // global
typedef __attribute__((address_space(3))) v4i v4i_l;   // LDS

#if defined(__HIP_DEVICE_COMPILE__) && \
    __has_builtin(__builtin_amdgcn_global_load_async_to_lds_b128) && \
    __has_builtin(__builtin_amdgcn_s_wait_asynccnt)
#define CDNA5_ASYNC 1
#else
#define CDNA5_ASYNC 0
#endif

// copy 16 bytes (8 bf16) global -> LDS
__device__ __forceinline__ void copy16(const bf16_t* __restrict__ g,
                                       bf16_t* __restrict__ l) {
#if CDNA5_ASYNC
  __builtin_amdgcn_global_load_async_to_lds_b128((v4i_g*)g, (v4i_l*)l, 0, 0);
#else
  *reinterpret_cast<v8bf*>(l) = *reinterpret_cast<const v8bf*>(g);
#endif
}

__device__ __forceinline__ void staging_wait() {
#if CDNA5_ASYNC
  __builtin_amdgcn_s_wait_asynccnt(0);
#endif
  __syncthreads();
}

// A-fragment (16x32 bf16): lane holds two contiguous 8-elem runs.
__device__ __forceinline__ v16bf load_afrag(const bf16_t* rowptr, int khalf) {
  v8bf lo = *reinterpret_cast<const v8bf*>(rowptr + khalf * 8);
  v8bf hi = *reinterpret_cast<const v8bf*>(rowptr + 16 + khalf * 8);
  return __builtin_shufflevector(lo, hi, 0, 1, 2, 3, 4, 5, 6, 7,
                                 8, 9, 10, 11, 12, 13, 14, 15);
}

// B-fragment (32x16 bf16) from an N-major (transposed) tile: lane holds two
// contiguous 8-elem runs at columns ksel*8 and 16+ksel*8.
__device__ __forceinline__ v16bf load_bfrag(const bf16_t* rowptr, int ksel) {
  v8bf lo = *reinterpret_cast<const v8bf*>(rowptr + ksel * 8);
  v8bf hi = *reinterpret_cast<const v8bf*>(rowptr + 16 + ksel * 8);
  return __builtin_shufflevector(lo, hi, 0, 1, 2, 3, 4, 5, 6, 7,
                                 8, 9, 10, 11, 12, 13, 14, 15);
}

// ---------------- fp32 [K,N] -> bf16 [N,K] transpose-cast ----------------
__global__ __launch_bounds__(256) void transpose_cast_kernel(
    const float* __restrict__ in, bf16_t* __restrict__ out, int K, int N) {
  __shared__ float tile[32][33];
  const int kt = blockIdx.y * 32, nt = blockIdx.x * 32;
  const int tx = threadIdx.x & 31, ty = threadIdx.x >> 5;  // 32 x 8
#pragma unroll
  for (int i = 0; i < 4; ++i)
    tile[ty + i * 8][tx] = in[(size_t)(kt + ty + i * 8) * N + nt + tx];
  __syncthreads();
#pragma unroll
  for (int i = 0; i < 4; ++i)
    out[(size_t)(nt + ty + i * 8) * K + kt + tx] = (bf16_t)tile[tx][ty + i * 8];
}

// ---------------- layernorm (unbiased var, scalar alpha/beta) ----------------
__global__ __launch_bounds__(256) void layernorm_bf16_kernel(
    const float* __restrict__ x, const float* __restrict__ alpha,
    const float* __restrict__ beta, bf16_t* __restrict__ y) {
  __shared__ float red[256];
  const int row = blockIdx.x;
  const int tid = threadIdx.x;
  const float* xr = x + (size_t)row * DD;

  float v[4];
  float sum = 0.f;
#pragma unroll
  for (int i = 0; i < 4; ++i) { v[i] = xr[tid + i * 256]; sum += v[i]; }
  red[tid] = sum;
  __syncthreads();
  for (int s = 128; s > 0; s >>= 1) {
    if (tid < s) red[tid] += red[tid + s];
    __syncthreads();
  }
  const float mean = red[0] / (float)DD;
  __syncthreads();

  float vs = 0.f;
#pragma unroll
  for (int i = 0; i < 4; ++i) { float d = v[i] - mean; vs += d * d; }
  red[tid] = vs;
  __syncthreads();
  for (int s = 128; s > 0; s >>= 1) {
    if (tid < s) red[tid] += red[tid + s];
    __syncthreads();
  }
  const float var = red[0] / (float)(DD - 1);
  const float inv = alpha[0] / (sqrtf(var) + 1e-6f);
  const float b0 = beta[0];

  bf16_t* yr = y + (size_t)row * DD;
#pragma unroll
  for (int i = 0; i < 4; ++i)
    yr[tid + i * 256] = (bf16_t)((v[i] - mean) * inv + b0);
}

// ---------------- bf16 WMMA GEMM: C = A[M,K] * Bt[N,K]^T + bias ----------------
#define BM 64
#define BN 256
#define BK 32

__global__ __launch_bounds__(256) void gemm_bf16_kernel(
    const bf16_t* __restrict__ A, const bf16_t* __restrict__ Bt,
    const float* __restrict__ bias, const float* __restrict__ resid,
    float* __restrict__ outf, bf16_t* __restrict__ outb,
    int M, int N, int K, int relu) {
  __shared__ __align__(16) bf16_t As[BM][BK + 8];        // 5.0 KB
  __shared__ __align__(16) bf16_t Bs[BN][BK + 8];        // 20.0 KB, N-major

  const int tid  = threadIdx.x;
  const int lane = tid & 31;
  const int wave = tid >> 5;    // 0..7
  const int wm   = wave & 1;    // M half (32 rows)
  const int wn   = wave >> 1;   // N quarter (64 cols)
  const int m0 = blockIdx.y * BM;
  const int n0 = blockIdx.x * BN;

  const int nrow  = lane & 15;
  const int khalf = lane >> 4;

  v8f acc[2][4];
#pragma unroll
  for (int m = 0; m < 2; ++m)
#pragma unroll
    for (int a = 0; a < 4; ++a)
#pragma unroll
      for (int j = 0; j < 8; ++j) acc[m][a][j] = 0.f;

  for (int k0 = 0; k0 < K; k0 += BK) {
    {  // stage A: 256 chunks of 8 bf16, one per thread
      int r = tid >> 2, c = (tid & 3) * 8;
      copy16(&A[(size_t)(m0 + r) * K + k0 + c], &As[r][c]);
    }
#pragma unroll
    for (int cc = tid; cc < (BN * BK) / 8; cc += 256) {  // 4 chunks per thread
      int n = cc >> 2, c = (cc & 3) * 8;
      copy16(&Bt[(size_t)(n0 + n) * K + k0 + c], &Bs[n][c]);
    }
    staging_wait();

#pragma unroll
    for (int m = 0; m < 2; ++m) {
      v16bf afrag = load_afrag(&As[wm * 32 + m * 16 + nrow][0], khalf);
#pragma unroll
      for (int a = 0; a < 4; ++a) {
        v16bf bfrag = load_bfrag(&Bs[wn * 64 + a * 16 + nrow][0], khalf);
        acc[m][a] = __builtin_amdgcn_wmma_f32_16x16x32_bf16(
            false, afrag, false, bfrag, (short)0, acc[m][a], false, false);
      }
    }
    __syncthreads();
  }

  const int hi = lane >> 4;
#pragma unroll
  for (int m = 0; m < 2; ++m) {
#pragma unroll
    for (int a = 0; a < 4; ++a) {
#pragma unroll
      for (int j = 0; j < 8; ++j) {
        int r = m0 + wm * 32 + m * 16 + j + 8 * hi;
        int c = n0 + wn * 64 + a * 16 + nrow;
        float v = acc[m][a][j] + bias[c];
        if (relu) v = v > 0.f ? v : 0.f;
        if (resid) v += resid[(size_t)r * N + c];
        if (outf) outf[(size_t)r * N + c] = v;
        if (outb) outb[(size_t)r * N + c] = (bf16_t)v;
      }
    }
  }
}

// ---------------- flash attention (online softmax) ----------------
#define QT  64   // q rows per block (4 waves x 16)
#define KVT 32   // kv columns per step

__global__ __launch_bounds__(128) void attention_kernel(
    const bf16_t* __restrict__ Qg, const bf16_t* __restrict__ Kg,
    const bf16_t* __restrict__ Vg, const int* __restrict__ mask,
    bf16_t* __restrict__ Og) {
  __shared__ __align__(16) bf16_t Qs[QT][DKK + 8];       // [q][dk]
  __shared__ __align__(16) bf16_t Ks[KVT][DKK + 8];      // [kv][dk]
  __shared__ __align__(16) bf16_t Vt[DKK][KVT + 8];      // [dk][kv] (transposed)
  __shared__ float  Ls[4][16][KVT];
  __shared__ __align__(16) bf16_t Ps[4][16][KVT + 8];
  __shared__ float  Sm[4][16], Sl[4][16], Sc[4][16];

  const int tid  = threadIdx.x;
  const int lane = tid & 31;
  const int wave = tid >> 5;       // 0..3
  const int bh = blockIdx.y;       // b*H + h
  const int b = bh / HH, h = bh % HH;
  const int q0 = blockIdx.x * QT;

  const size_t rowbase = (size_t)b * SS;
  const int colbase = h * DKK;

  // stage Q tile: 512 chunks of 8 bf16, 4 per thread
#pragma unroll
  for (int cc = tid; cc < (QT * DKK) / 8; cc += 128) {
    int r = cc >> 3, c = (cc & 7) * 8;
    copy16(&Qg[(rowbase + q0 + r) * DD + colbase + c], &Qs[r][c]);
  }
  if (lane < 16) { Sm[wave][lane] = -3.0e38f; Sl[wave][lane] = 0.f; }

  v8f o[4];
#pragma unroll
  for (int a = 0; a < 4; ++a)
#pragma unroll
    for (int j = 0; j < 8; ++j) o[a][j] = 0.f;

  const int nrow = lane & 15;
  const int ksel = lane >> 4;
  staging_wait();

  for (int kv = 0; kv < SS; kv += KVT) {
    // stage K chunk [kv][dk]: 256 chunks, 2 per thread
#pragma unroll
    for (int cc = tid; cc < (KVT * DKK) / 8; cc += 128) {
      int r = cc >> 3, c = (cc & 7) * 8;
      copy16(&Kg[(rowbase + kv + r) * DD + colbase + c], &Ks[r][c]);
    }
    // stage V transposed [dk][kv]: vector global read, scalar LDS scatter
#pragma unroll
    for (int cc = tid; cc < (KVT * DKK) / 8; cc += 128) {
      int r = cc >> 3, c = (cc & 7) * 8;
      v8bf g = *reinterpret_cast<const v8bf*>(
          &Vg[(rowbase + kv + r) * DD + colbase + c]);
#pragma unroll
      for (int e = 0; e < 8; ++e) Vt[c + e][r] = g[e];
    }
    staging_wait();

    // logits(16x32) = Qtile(16x64) x K^T(64x32)
    v8f s[2];
#pragma unroll
    for (int t = 0; t < 2; ++t)
#pragma unroll
      for (int j = 0; j < 8; ++j) s[t][j] = 0.f;

#pragma unroll
    for (int kc = 0; kc < 2; ++kc) {
      v16bf afrag = load_afrag(&Qs[wave * 16 + nrow][kc * 32], ksel);
#pragma unroll
      for (int t = 0; t < 2; ++t) {
        // B(k=dk, n=kv): contiguous over dk in kv-major Ks
        v16bf bfrag = load_bfrag(&Ks[t * 16 + nrow][kc * 32], ksel);
        s[t] = __builtin_amdgcn_wmma_f32_16x16x32_bf16(
            false, afrag, false, bfrag, (short)0, s[t], false, false);
      }
    }

    // dump scaled+masked logits to per-wave LDS scratch
#pragma unroll
    for (int t = 0; t < 2; ++t)
#pragma unroll
      for (int j = 0; j < 8; ++j) {
        int r = j + 8 * ksel;
        int c = t * 16 + nrow;
        float v = s[t][j] * 0.125f;  // 1/sqrt(64)
        if (mask[b * SS + kv + c] == 0) v = -1e9f;
        Ls[wave][r][c] = v;
      }

    // online softmax update: lanes 0..15 each own one q row
    if (lane < 16) {
      int r = lane;
      float mprev = Sm[wave][r];
      float tmax = -3.0e38f;
      for (int c = 0; c < KVT; ++c) tmax = fmaxf(tmax, Ls[wave][r][c]);
      float mnew = fmaxf(mprev, tmax);
      float corr = __expf(mprev - mnew);
      float sum = 0.f;
      for (int c = 0; c < KVT; ++c) {
        float p = __expf(Ls[wave][r][c] - mnew);
        Ps[wave][r][c] = (bf16_t)p;
        sum += p;
      }
      Sl[wave][r] = Sl[wave][r] * corr + sum;
      Sm[wave][r] = mnew;
      Sc[wave][r] = corr;
    }

    // rescale running output by per-row correction
#pragma unroll
    for (int a = 0; a < 4; ++a)
#pragma unroll
      for (int j = 0; j < 8; ++j)
        o[a][j] *= Sc[wave][j + 8 * ksel];

    // o(16x64) += P(16x32) x V(32x64)
    v16bf pfrag = load_afrag(&Ps[wave][nrow][0], ksel);
#pragma unroll
    for (int a = 0; a < 4; ++a) {
      // B(k=kv, n=dk): contiguous over kv in dk-major Vt
      v16bf vfrag = load_bfrag(&Vt[a * 16 + nrow][0], ksel);
      o[a] = __builtin_amdgcn_wmma_f32_16x16x32_bf16(
          false, pfrag, false, vfrag, (short)0, o[a], false, false);
    }
    __syncthreads();
  }

  // normalize and write [B,S,H,DK] == [B,S,D]
#pragma unroll
  for (int a = 0; a < 4; ++a)
#pragma unroll
    for (int j = 0; j < 8; ++j) {
      int r = j + 8 * ksel;
      float v = o[a][j] / Sl[wave][r];
      int gr = q0 + wave * 16 + r;
      int gc = colbase + a * 16 + nrow;
      Og[(rowbase + gr) * DD + gc] = (bf16_t)v;
    }
}

// ---------------- host launcher ----------------
extern "C" void kernel_launch(void* const* d_in, const int* in_sizes, int n_in,
                              void* d_out, int out_size, void* d_ws, size_t ws_size,
                              hipStream_t stream) {
  (void)in_sizes; (void)n_in; (void)out_size; (void)ws_size;
  const float* xb   = (const float*)d_in[0];
  const int*   mask = (const int*)d_in[1];
  const float* Wq = (const float*)d_in[2];  const float* bq = (const float*)d_in[3];
  const float* Wk = (const float*)d_in[4];  const float* bk = (const float*)d_in[5];
  const float* Wv = (const float*)d_in[6];  const float* bv = (const float*)d_in[7];
  const float* Wo = (const float*)d_in[8];  const float* bo = (const float*)d_in[9];
  const float* W1 = (const float*)d_in[10]; const float* b1 = (const float*)d_in[11];
  const float* W2 = (const float*)d_in[12]; const float* b2 = (const float*)d_in[13];
  const float* a1 = (const float*)d_in[14]; const float* o1 = (const float*)d_in[15];
  const float* a2 = (const float*)d_in[16]; const float* o2 = (const float*)d_in[17];

  constexpr size_t MB = 1024ull * 1024ull;
  char* ws = (char*)d_ws;
  bf16_t* WqT = (bf16_t*)(ws + 0 * MB);      // [N,K] bf16, 2MB each
  bf16_t* WkT = (bf16_t*)(ws + 2 * MB);
  bf16_t* WvT = (bf16_t*)(ws + 4 * MB);
  bf16_t* WoT = (bf16_t*)(ws + 6 * MB);
  bf16_t* W1T = (bf16_t*)(ws + 8 * MB);      // [DFF,D] bf16, 8MB
  bf16_t* W2T = (bf16_t*)(ws + 16 * MB);     // [D,DFF] bf16, 8MB
  bf16_t* XLN = (bf16_t*)(ws + 24 * MB);     // [4096,1024] bf16 (reused)
  bf16_t* Qb  = (bf16_t*)(ws + 32 * MB);
  bf16_t* Kb  = (bf16_t*)(ws + 40 * MB);
  bf16_t* Vb  = (bf16_t*)(ws + 48 * MB);
  bf16_t* AOb = (bf16_t*)(ws + 56 * MB);
  float*  Hf  = (float*)(ws + 64 * MB);      // [4096,1024] f32, 16MB
  bf16_t* FFH = (bf16_t*)(ws + 80 * MB);     // [4096,4096] bf16, 32MB

  // 1) weights -> bf16, transposed to [N,K]
  transpose_cast_kernel<<<dim3(DD/32, DD/32), 256, 0, stream>>>(Wq, WqT, DD, DD);
  transpose_cast_kernel<<<dim3(DD/32, DD/32), 256, 0, stream>>>(Wk, WkT, DD, DD);
  transpose_cast_kernel<<<dim3(DD/32, DD/32), 256, 0, stream>>>(Wv, WvT, DD, DD);
  transpose_cast_kernel<<<dim3(DD/32, DD/32), 256, 0, stream>>>(Wo, WoT, DD, DD);
  transpose_cast_kernel<<<dim3(DFFF/32, DD/32), 256, 0, stream>>>(W1, W1T, DD, DFFF);
  transpose_cast_kernel<<<dim3(DD/32, DFFF/32), 256, 0, stream>>>(W2, W2T, DFFF, DD);

  // 2) LN1(xb) -> bf16
  layernorm_bf16_kernel<<<MROWS, 256, 0, stream>>>(xb, a1, o1, XLN);

  // 3) Q/K/V projections
  dim3 gD(DD / BN, MROWS / BM);
  gemm_bf16_kernel<<<gD, 256, 0, stream>>>(XLN, WqT, bq, nullptr, nullptr, Qb,
                                           MROWS, DD, DD, 0);
  gemm_bf16_kernel<<<gD, 256, 0, stream>>>(XLN, WkT, bk, nullptr, nullptr, Kb,
                                           MROWS, DD, DD, 0);
  gemm_bf16_kernel<<<gD, 256, 0, stream>>>(XLN, WvT, bv, nullptr, nullptr, Vb,
                                           MROWS, DD, DD, 0);

  // 4) attention
  attention_kernel<<<dim3(SS / QT, BB * HH), 128, 0, stream>>>(Qb, Kb, Vb, mask, AOb);

  // 5) output projection + residual: h = xb + AO @ Wo + bo
  gemm_bf16_kernel<<<gD, 256, 0, stream>>>(AOb, WoT, bo, xb, Hf, nullptr,
                                           MROWS, DD, DD, 0);

  // 6) LN2(h) -> bf16
  layernorm_bf16_kernel<<<MROWS, 256, 0, stream>>>(Hf, a2, o2, XLN);

  // 7) FFN
  dim3 gF(DFFF / BN, MROWS / BM);
  gemm_bf16_kernel<<<gF, 256, 0, stream>>>(XLN, W1T, b1, nullptr, nullptr, FFH,
                                           MROWS, DFFF, DD, 1);
  gemm_bf16_kernel<<<gD, 256, 0, stream>>>(FFH, W2T, b2, Hf, (float*)d_out, nullptr,
                                           MROWS, DD, DFFF, 0);
}
